// Downsample_IS_29368986370262
// MI455X (gfx1250) — matrix operand order repaired and checked by reading, not compiled, not run on hardware
//
#include <hip/hip_runtime.h>
#include <hip/hip_bf16.h>

typedef __attribute__((ext_vector_type(16))) _Float16 v16h;
typedef __attribute__((ext_vector_type(8)))  _Float16 v8h;
typedef __attribute__((ext_vector_type(8)))  float    v8f;

#define BN     8
#define CIN    64
#define COUT   32
#define HW     256
#define HO     254      // valid-conv output height/width
#define KTOT   576      // CIN * 9
#define CSTR   72       // channel stride in LDS x tile (16B-aligned rows)

// workspace byte offsets
#define OFF_W16   0          // 18432 f16 (K-permuted) -> 36864 B
#define OFF_SUMS  36864      // 8*32 f32              ->  1024 B
#define OFF_KERN  37888      // 8*9  f32              ->   288 B

// ---------------------------------------------------------------------------
// Kernel 0: permute conv weights to K' = (kh*3+kw)*64 + c ordering (tap-major,
// channel-minor) as f16, and zero the mean accumulators (ws is poisoned
// before timing, so this must run every call).
// ---------------------------------------------------------------------------
__global__ __launch_bounds__(256) void prep_kernel(const float* __restrict__ w,
                                                   _Float16* __restrict__ w16p,
                                                   float* __restrict__ sums) {
    int idx = blockIdx.x * 256 + threadIdx.x;
    if (idx < COUT * KTOT) {
        int o    = idx / KTOT;
        int kp   = idx - o * KTOT;      // K' = tap*64 + c
        int tap  = kp >> 6;
        int c    = kp & 63;
        w16p[idx] = (_Float16)w[o * KTOT + c * 9 + tap];
    }
    if (idx < BN * COUT) sums[idx] = 0.f;
}

// ---------------------------------------------------------------------------
// Kernel 1: implicit-GEMM 3x3 conv (64->32) with f16 WMMA + relu + spatial
// mean reduction. Block = (b, out-row h, 64-wide w segment), 8 waves.
// Wave wid: M-tile mt = wid&3 (16 consecutive w positions), N-tile nt = wid>>2.
// K' chunks: chunk ck covers tap = ck>>1 (compile-time) and channels
// (ck&1)*32..+32, so all fragment addresses are perLaneBase + constant(ck).
// ---------------------------------------------------------------------------
__global__ __launch_bounds__(256) void conv_reduce_kernel(
    const float* __restrict__ x, const _Float16* __restrict__ w16p,
    float* __restrict__ sums) {

    // x tile: [row = r*66 + w][CSTR channels], 3*66*72 f16 = 28512 B
    __shared__ __attribute__((aligned(16))) _Float16 lds_x[3 * 66 * CSTR];
    // weights: [cout][K'] packed, 32*576 f16 = 36864 B (rows 16B aligned)
    __shared__ __attribute__((aligned(16))) _Float16 lds_w[COUT * KTOT];
    __shared__ float lds_sums[COUT];

    const int tid  = threadIdx.x;
    const int seg  = blockIdx.x & 3;
    int rest       = blockIdx.x >> 2;
    const int h    = rest % HO;
    const int b    = rest / HO;
    const int wseg = seg * 64;

    // Stage x tile: read coalesced in (c, r, w) order, scatter to
    // channel-minor LDS layout (f32 -> f16 convert on the fly).
    for (int i = tid; i < CIN * 3 * 66; i += 256) {
        int dw = i % 66;
        int t  = i / 66;          // t = c*3 + r
        int r  = t % 3;
        int c  = t / 3;
        int gw = wseg + dw;
        float v = (gw < HW) ? x[((b * CIN + c) * HW + (h + r)) * HW + gw] : 0.f;
        lds_x[(r * 66 + dw) * CSTR + c] = (_Float16)v;
    }
    // Stage pre-permuted weights: straight 16B copies (36864 B = 2304 uint4).
    {
        const uint4* src = (const uint4*)w16p;
        uint4*       dst = (uint4*)lds_w;
        for (int i = tid; i < (COUT * KTOT) / 8; i += 256) dst[i] = src[i];
    }
    if (tid < COUT) lds_sums[tid] = 0.f;
    __syncthreads();

    const int lane  = tid & 31;
    const int wid   = tid >> 5;
    const int mt    = wid & 3;
    const int nt    = wid >> 2;
    const int m     = lane & 15;
    const int hi    = lane >> 4;                 // lane half
    const int abase = (mt * 16 + m) * CSTR + hi * 16;       // per-lane A base
    const int bbase = (lane & 15) * KTOT + hi * 16;         // per-lane B base

    v8f acc = {};
#pragma unroll
    for (int ck = 0; ck < 18; ++ck) {            // K' = 576 in chunks of 32
        const int tap   = ck >> 1;               // compile-time after unroll
        const int r     = tap / 3;
        const int kw    = tap - r * 3;
        const int cbase = (ck & 1) * 32;
        const int ca    = (r * 66 + kw) * CSTR + cbase;     // A const offset
        // A fragment: elems 0..15 = channels cbase+hi*16+0..15 (contiguous).
        v8h a0 = *reinterpret_cast<const v8h*>(&lds_x[abase + ca]);
        v8h a1 = *reinterpret_cast<const v8h*>(&lds_x[abase + ca + 8]);
        v16h a = __builtin_shufflevector(a0, a1,
                 0, 1, 2, 3, 4, 5, 6, 7, 8, 9, 10, 11, 12, 13, 14, 15);
        // B fragment: one contiguous 32 B read.
        const v16h bf = *reinterpret_cast<const v16h*>(&lds_w[bbase + ck * 32]);

        acc = __builtin_amdgcn_wmma_f32_16x16x32_f16(
            false, a, false, bf, (short)0, acc, false, false);
    }

    // relu + sum over valid M rows.
    // C layout: lane l holds N = l&15, rows (l<16 ? i : i+8) for vgpr i.
    const int wbase = wseg + mt * 16 + hi * 8;   // w of acc[0]'s row
    float partial = 0.f;
#pragma unroll
    for (int i = 0; i < 8; ++i) {
        float v = fmaxf(acc[i], 0.f);
        partial += (wbase + i < HO) ? v : 0.f;
    }
    partial += __shfl_xor(partial, 16, 32);
    if (lane < 16) atomicAdd(&lds_sums[nt * 16 + lane], partial);
    __syncthreads();
    if (tid < COUT) atomicAdd(&sums[b * COUT + tid], lds_sums[tid]);
}

// ---------------------------------------------------------------------------
// Kernel 2: mean -> fc -> sigmoid -> 3x3 normalized Gaussian taps.
// One wave32 per sample (8 waves in one block).
// ---------------------------------------------------------------------------
__global__ __launch_bounds__(256) void head_kernel(const float* __restrict__ sums,
                                                   const float* __restrict__ w_fc,
                                                   const float* __restrict__ b_fc,
                                                   float* __restrict__ kern) {
    const int lane = threadIdx.x & 31;
    const int b    = threadIdx.x >> 5;
    const float inv_area = 1.f / (254.f * 254.f);
    float v = sums[b * COUT + lane] * inv_area * w_fc[lane];
#pragma unroll
    for (int off = 16; off > 0; off >>= 1) v += __shfl_xor(v, off, 32);
    if (lane == 0) {
        float z   = v + b_fc[0];
        float sig = 1.f / (1.f + __expf(-z));
        float s   = fmaxf(sig, 1e-4f);
        float is2 = 1.f / (s * s);
        float e1  = __expf(-0.5f * is2);   // d = 1
        float e2  = e1 * e1;               // d = sqrt(2)
        float nrm = 1.f / (1.f + 4.f * e1 + 4.f * e2);
#pragma unroll
        for (int kh = 0; kh < 3; ++kh)
#pragma unroll
            for (int kw = 0; kw < 3; ++kw) {
                int d2 = (kh - 1) * (kh - 1) + (kw - 1) * (kw - 1);
                float g = (d2 == 0) ? 1.f : ((d2 == 1) ? e1 : e2);
                kern[b * 9 + kh * 3 + kw] = g * nrm;
            }
    }
}

// ---------------------------------------------------------------------------
// Kernel 3: per-sample 3x3 Gaussian blur, stride 2, reflect pad.
// out[c][b][ho][wo]; one thread per output element (bandwidth bound, x is
// L2-resident at 134 MB < 192 MB).
// ---------------------------------------------------------------------------
__global__ __launch_bounds__(256) void blur_kernel(const float* __restrict__ x,
                                                   const float* __restrict__ kern,
                                                   float* __restrict__ out) {
    int idx = blockIdx.x * 256 + threadIdx.x;
    int wo = idx & 127;
    int ho = (idx >> 7) & 127;
    int b  = (idx >> 14) & 7;
    int c  = idx >> 17;
    const float* xp = x + ((size_t)(b * CIN + c)) * HW * HW;
    const float* kb = kern + b * 9;
    float acc = 0.f;
#pragma unroll
    for (int kh = 0; kh < 3; ++kh) {
        int iy = 2 * ho - 1 + kh;
        iy = iy < 0 ? -iy : (iy > HW - 1 ? 2 * (HW - 1) - iy : iy);
#pragma unroll
        for (int kw = 0; kw < 3; ++kw) {
            int ix = 2 * wo - 1 + kw;
            ix = ix < 0 ? -ix : (ix > HW - 1 ? 2 * (HW - 1) - ix : ix);
            acc += kb[kh * 3 + kw] * xp[iy * HW + ix];
        }
    }
    out[idx] = acc;
}

// ---------------------------------------------------------------------------
extern "C" void kernel_launch(void* const* d_in, const int* in_sizes, int n_in,
                              void* d_out, int out_size, void* d_ws, size_t ws_size,
                              hipStream_t stream) {
    const float* x       = (const float*)d_in[0];
    const float* w_sigma = (const float*)d_in[1];
    const float* w_fc    = (const float*)d_in[2];
    const float* b_fc    = (const float*)d_in[3];

    char* ws = (char*)d_ws;
    _Float16* w16p = (_Float16*)(ws + OFF_W16);
    float* sums    = (float*)(ws + OFF_SUMS);
    float* kern    = (float*)(ws + OFF_KERN);
    float* out     = (float*)d_out;

    // 0) weights -> f16 with K' permutation, zero accumulators
    prep_kernel<<<72, 256, 0, stream>>>(w_sigma, w16p, sums);
    // 1) WMMA conv + relu + spatial-mean: 8 samples * 254 rows * 4 segments
    conv_reduce_kernel<<<BN * HO * 4, 256, 0, stream>>>(x, w16p, sums);
    // 2) fc + sigmoid + gaussian taps
    head_kernel<<<1, 256, 0, stream>>>(sums, w_fc, b_fc, kern);
    // 3) blur: 64*8*128*128 outputs / 256
    blur_kernel<<<(CIN * BN * 128 * 128) / 256, 256, 0, stream>>>(x, kern, out);
}
